// MultiHeadedAttention_64390149702100
// MI455X (gfx1250) — matrix operand order repaired and checked by reading, not compiled, not run on hardware
//
#include <hip/hip_runtime.h>
#include <stdint.h>

typedef __attribute__((ext_vector_type(16))) __bf16 v16bf;
typedef __attribute__((ext_vector_type(8)))  float  v8f;

#define TOKENS   (16 * 4096)     // B*S = 65536
#define DM       512             // model dim (K of the GEMMs)
#define NH       512             // H*D   (N of each GEMM)
#define NHEADS   8
#define HDIM     64
#define MTILE    64              // tokens per block
#define KCHUNKS  (DM / 32)       // 16
#define NTILES   (NH / 16)       // 32

__device__ __forceinline__ unsigned short f32_to_bf16(float f) {
    unsigned int x = __float_as_uint(f);
    x += 0x7FFFu + ((x >> 16) & 1u);            // round-to-nearest-even
    return (unsigned short)(x >> 16);
}
__device__ __forceinline__ float bf16lo(unsigned int u) { return __uint_as_float(u << 16); }
__device__ __forceinline__ float bf16hi(unsigned int u) { return __uint_as_float(u & 0xFFFF0000u); }

// ---------------------------------------------------------------------------
// Kernel 0: repack fp32 weights (Wq|Wk|Wv, each 512x512 row-major K x N) into
// the WMMA B-fragment-native bf16 layout:
//   frag index = ((m*16 + kc)*32 + nt)*32 + lane, 16 bf16 per frag where
//   component j holds W[kc*32 + (lane>>4)*8 + (j&7) + (j>>3)*16][nt*16 + (lane&15)].
// Each lane's fragment is a contiguous 32-byte chunk -> 2 coalesced b128 loads.
// ---------------------------------------------------------------------------
__global__ void pack_weights(const float* __restrict__ wq,
                             const float* __restrict__ wk,
                             const float* __restrict__ wv,
                             unsigned short* __restrict__ wp) {
    unsigned idx  = blockIdx.x * 256u + threadIdx.x;   // < 3 * 2^18
    unsigned j    =  idx        & 15u;
    unsigned lane = (idx >> 4)  & 31u;
    unsigned nt   = (idx >> 9)  & 31u;
    unsigned kc   = (idx >> 14) & 15u;
    unsigned m    =  idx >> 18;
    const float* w = (m == 0u) ? wq : (m == 1u) ? wk : wv;
    unsigned k = kc * 32u + (lane >> 4) * 8u + (j & 7u) + (j >> 3) * 16u;
    unsigned n = nt * 16u + (lane & 15u);
    wp[idx] = f32_to_bf16(w[k * DM + n]);
}

// ---------------------------------------------------------------------------
// Kernel 1: fused QKV projection (bf16 WMMA, fp32 accum) + cross-head
// attention per token. One block = 64 tokens, 8 waves (4 row-stripes x 2
// column halves). Q/K/V staged in 192 KB of LDS as bf16.
// ---------------------------------------------------------------------------
__global__ void __launch_bounds__(256)
fused_qkv_attn(const float* __restrict__ x,
               const unsigned short* __restrict__ wp,
               float* __restrict__ out) {
    extern __shared__ unsigned short qkv[];   // [3][MTILE][NH] bf16 bits

    const int tid  = threadIdx.x;
    const int lane = tid & 31;
    const int wave = tid >> 5;
    const int rowStripe = wave & 3;           // which 16-row stripe
    const int colHalf   = wave >> 2;          // which 256-col half
    const int laneRow = lane & 15;
    const int laneH   = lane >> 4;
    const int rowBase = blockIdx.x * MTILE;

    // A-fragment source row for this lane (16-bit A 16x32 ISA layout)
    const int r = rowBase + rowStripe * 16 + laneRow;
    const float4* xrow = (const float4*)(x + (size_t)r * DM);

    for (int m = 0; m < 3; ++m) {
        v8f zero = {0.f, 0.f, 0.f, 0.f, 0.f, 0.f, 0.f, 0.f};
        v8f acc[16];
        #pragma unroll
        for (int t = 0; t < 16; ++t) acc[t] = zero;

        for (int kc = 0; kc < KCHUNKS; ++kc) {
            // ---- A fragment: lane half h holds K = {h*8..h*8+7, 16+h*8..+7}
            const int f4 = kc * 8 + laneH * 2;
            const float4 a0 = xrow[f4 + 0];
            const float4 a1 = xrow[f4 + 1];
            const float4 a2 = xrow[f4 + 4];
            const float4 a3 = xrow[f4 + 5];
            v16bf af;
            af[0]  = (__bf16)a0.x; af[1]  = (__bf16)a0.y; af[2]  = (__bf16)a0.z; af[3]  = (__bf16)a0.w;
            af[4]  = (__bf16)a1.x; af[5]  = (__bf16)a1.y; af[6]  = (__bf16)a1.z; af[7]  = (__bf16)a1.w;
            af[8]  = (__bf16)a2.x; af[9]  = (__bf16)a2.y; af[10] = (__bf16)a2.z; af[11] = (__bf16)a2.w;
            af[12] = (__bf16)a3.x; af[13] = (__bf16)a3.y; af[14] = (__bf16)a3.z; af[15] = (__bf16)a3.w;

            // ---- B fragments: contiguous 32B per lane from packed weights
            const v16bf* bbase = (const v16bf*)wp +
                (size_t)(((m * KCHUNKS + kc) * NTILES + colHalf * 16) * 32 + lane);
            #pragma unroll
            for (int t = 0; t < 16; ++t) {
                v16bf bf = bbase[t * 32];
                acc[t] = __builtin_amdgcn_wmma_f32_16x16x32_bf16(
                    false, af, false, bf, (short)0, acc[t], false, false);
            }
            if (kc + 1 < KCHUNKS)
                __builtin_prefetch((const void*)(bbase + 32 * 32), 0, 0);
        }

        // ---- park this matrix's 64x512 result in LDS as bf16
        unsigned short* qs = qkv + m * (MTILE * NH);
        #pragma unroll
        for (int t = 0; t < 16; ++t) {
            const int col = (colHalf * 16 + t) * 16 + laneRow;   // C: N = lane%16
            #pragma unroll
            for (int v = 0; v < 8; ++v) {
                const int rl = rowStripe * 16 + laneH * 8 + v;   // C: M = (lane/16)*8 + v
                qs[rl * NH + col] = f32_to_bf16(acc[t][v]);
            }
        }
    }
    __syncthreads();

    // ---- Phase 2: per-token cross-head attention (H x H scores, softmax, AV)
    const unsigned int* q32 = (const unsigned int*)qkv;  // packed bf16 pairs
    for (int pp = 0; pp < 2; ++pp) {
        const int pi  = tid + pp * 256;       // (token, head) pair id
        const int tok = pi >> 3;
        const int h   = pi & 7;
        const int qb  = tok * 256 + h * 32;   // uint32 index into Q plane

        float s[NHEADS];
        #pragma unroll
        for (int t2 = 0; t2 < NHEADS; ++t2) {
            const int kb = 16384 + tok * 256 + t2 * 32;   // K plane
            float a = 0.f;
            #pragma unroll
            for (int dw = 0; dw < 32; ++dw) {
                const unsigned int qw = q32[qb + dw];
                const unsigned int kw = q32[kb + dw];
                a += bf16lo(qw) * bf16lo(kw) + bf16hi(qw) * bf16hi(kw);
            }
            s[t2] = a;
        }
        float mx = s[0];
        #pragma unroll
        for (int t2 = 1; t2 < NHEADS; ++t2) mx = fmaxf(mx, s[t2]);
        float sum = 0.f;
        #pragma unroll
        for (int t2 = 0; t2 < NHEADS; ++t2) { s[t2] = __expf(s[t2] - mx); sum += s[t2]; }
        const float inv = 1.0f / sum;
        #pragma unroll
        for (int t2 = 0; t2 < NHEADS; ++t2) s[t2] *= inv;

        float2* op = (float2*)(out + (size_t)(rowBase + tok) * NH + h * HDIM);
        #pragma unroll
        for (int dw = 0; dw < 32; ++dw) {
            float o0 = 0.f, o1 = 0.f;
            #pragma unroll
            for (int t2 = 0; t2 < NHEADS; ++t2) {
                const unsigned int vw = q32[32768 + tok * 256 + t2 * 32 + dw]; // V plane
                o0 += s[t2] * bf16lo(vw);
                o1 += s[t2] * bf16hi(vw);
            }
            op[dw] = make_float2(o0, o1);
        }
    }
}

extern "C" void kernel_launch(void* const* d_in, const int* in_sizes, int n_in,
                              void* d_out, int out_size, void* d_ws, size_t ws_size,
                              hipStream_t stream) {
    const float* x  = (const float*)d_in[0];
    const float* wq = (const float*)d_in[1];
    const float* wk = (const float*)d_in[2];
    const float* wv = (const float*)d_in[3];
    (void)in_sizes; (void)n_in; (void)out_size; (void)ws_size;

    unsigned short* wp = (unsigned short*)d_ws;   // 3*512*512*2 = 1.5 MB of scratch

    // Repack weights to WMMA-B-native bf16 layout (3*2^18 elements).
    pack_weights<<<3072, 256, 0, stream>>>(wq, wk, wv, wp);

    // Fused projection + attention: 1024 blocks x 256 threads, 192 KB LDS.
    const size_t smem = (size_t)3 * MTILE * NH * sizeof(unsigned short); // 196608 B
    fused_qkv_attn<<<TOKENS / MTILE, 256, smem, stream>>>(x, wp, (float*)d_out);
}
